// LinformerFusion_47029891891840
// MI455X (gfx1250) — matrix operand order repaired and checked by reading, not compiled, not run on hardware
//
#include <hip/hip_runtime.h>
#include <math.h>

typedef __attribute__((ext_vector_type(16))) __bf16 v16bf;
typedef __attribute__((ext_vector_type(8)))  float  v8f;

#define BDIM 8
#define CCH  96
#define HH   96
#define WW   96
#define NTOK (HH*WW)        // 9216
#define DIMF 192
#define NHEADS 4
#define DH   48
#define DHP  64             // padded head dim
#define KP   256
#define FFD  768

static __device__ __forceinline__ unsigned short f2bf(float f) {
  unsigned int u = __float_as_uint(f);
  u += 0x7FFFu + ((u >> 16) & 1u);          // round-to-nearest-even
  return (unsigned short)(u >> 16);
}

// ---------------------------------------------------------------- utility kernels
__global__ void fill_zero_u32(unsigned int* __restrict__ p, long n) {
  long i = (long)blockIdx.x * blockDim.x + threadIdx.x;
  if (i < n) p[i] = 0u;
}
__global__ void cvt_f32_bf16(const float* __restrict__ s, unsigned short* __restrict__ d, long n) {
  long i = (long)blockIdx.x * blockDim.x + threadIdx.x;
  if (i < n) d[i] = f2bf(s[i]);
}
// s: R x Cc row-major (f32) -> d: Cc x R row-major (bf16)
__global__ void cvt_transpose_bf16(const float* __restrict__ s, unsigned short* __restrict__ d,
                                   int R, int Cc) {
  long i = (long)blockIdx.x * blockDim.x + threadIdx.x;
  if (i < (long)R * Cc) {
    int r = (int)(i / Cc);
    int c = (int)(i - (long)r * Cc);
    d[(long)c * R + r] = f2bf(s[i]);
  }
}

// ---------------------------------------------------------------- pack / LN
__global__ void pack_x(const float* __restrict__ detail, const float* __restrict__ smooth,
                       float* __restrict__ xf) {
  int tok = blockIdx.x;                 // b*NTOK + n
  int d   = threadIdx.x;                // 0..191
  int b   = tok / NTOK;
  int n   = tok - b * NTOK;
  int hi  = n / WW, wi = n - hi * WW;
  float v = (d < CCH) ? detail[(((long)b*CCH + d)      *HH + hi)*WW + wi]
                      : smooth[(((long)b*CCH + (d-CCH))*HH + hi)*WW + wi];
  xf[(long)tok * DIMF + d] = v;
}

__global__ __launch_bounds__(256)
void layernorm_bf16(const float* __restrict__ xin, const float* __restrict__ gam,
                    const float* __restrict__ bet, unsigned short* __restrict__ outp) {
  __shared__ float s1[256], s2[256];
  int tok = blockIdx.x;
  int d   = threadIdx.x;
  float v = 0.f;
  if (d < DIMF) v = xin[(long)tok * DIMF + d];
  s1[d] = v; s2[d] = v * v;
  __syncthreads();
  for (int st = 128; st > 0; st >>= 1) {
    if (d < st) { s1[d] += s1[d + st]; s2[d] += s2[d + st]; }
    __syncthreads();
  }
  float mean = s1[0] * (1.0f / DIMF);
  float var  = s2[0] * (1.0f / DIMF) - mean * mean;
  float rs   = rsqrtf(var + 1e-5f);
  if (d < DIMF) outp[(long)tok * DIMF + d] = f2bf((v - mean) * rs * gam[d] + bet[d]);
}

// ---------------------------------------------------------------- softmax
__global__ __launch_bounds__(256)
void softmax_row(const float* __restrict__ dots, unsigned short* __restrict__ attn) {
  __shared__ float sm[256];
  long row = blockIdx.x;
  int  c   = threadIdx.x;
  float v  = dots[row * KP + c] * 0.14433756729740645f;  // 1/sqrt(48)
  sm[c] = v; __syncthreads();
  for (int st = 128; st > 0; st >>= 1) {
    if (c < st) sm[c] = fmaxf(sm[c], sm[c + st]);
    __syncthreads();
  }
  float mx = sm[0]; __syncthreads();
  float e = expf(v - mx);
  sm[c] = e; __syncthreads();
  for (int st = 128; st > 0; st >>= 1) {
    if (c < st) sm[c] += sm[c + st];
    __syncthreads();
  }
  attn[row * KP + c] = f2bf(e / sm[0]);
}

// ---------------------------------------------------------------- WMMA GEMM
// D(MxN) = A(MxKd) * B [+bias][+res][gelu].  Block tile 128x64, BK=32.
// 8 waves as 4x2; each wave owns 32x32 -> 4 x v_wmma_f32_16x16x32_bf16 per K-step.
// Requirements (no load guards): M % 128 == 0, Kd % 32 == 0, operand buffers fully
// materialized over the addressed range. Epilogue guards columns (N may be ragged).
enum { F_TRANSB = 1, F_OUTBF16 = 2, F_BIAS = 4, F_RES = 8, F_GELU = 16, F_SPLITH = 32 };

template<int FLAGS>
__global__ __launch_bounds__(256)
void gemm_wmma(const unsigned short* __restrict__ A,
               const unsigned short* __restrict__ Bsrc,
               const float* __restrict__ bias,
               const float* __restrict__ res,
               float* __restrict__ Df,
               unsigned short* __restrict__ Dh,
               int M, int Nn, int Kd,
               int lda, int ldb, int ldd, int heads,
               long aSB, long aSH, long bSB, long bSH, long dSB, long dSH) {
  __shared__ unsigned short As[128 * 40];  // [m][k] row-major, stride 40
  __shared__ unsigned short Bs[64 * 40];   // [n][k] col-major(k-contig), stride 40

  const int t    = threadIdx.x;
  const int lane = t & 31;
  const int wave = t >> 5;
  const int wm   = wave >> 1;          // 0..3 (32-row group)
  const int wn   = wave & 1;           // 0..1 (32-col group)
  const int half = lane >> 4;
  const int l15  = lane & 15;

  const int z  = blockIdx.z;
  const int bb = z / heads;
  const int hh = z - bb * heads;
  const long aOff = (long)bb * aSB + (long)hh * aSH;
  const long bOff = (long)bb * bSB + (long)hh * bSH;
  const long dOff = (long)bb * dSB + (long)hh * dSH;

  const int rowBase = blockIdx.y * 128;
  const int colBase = blockIdx.x * 64;

  v8f acc00 = {}, acc01 = {}, acc10 = {}, acc11 = {};

  // global->LDS coordinates
  const int ar  = t >> 1;              // 0..127  (A row)
  const int ak  = (t & 1) << 4;        // 0 / 16  (A k-base, 16 elems per thread)
  const int bnr = t >> 2;              // 0..63   (TRANSB: B row = n)
  const int bkc = (t & 3) << 3;        // 0..24
  const int bkr = t >> 3;              // 0..31   (normal B: k row)
  const int bnc = (t & 7) << 3;        // 0..56

  // clamp for ragged-N TRANSB sources (branch-free; bogus rows masked at store)
  int bnrClamp = colBase + bnr;
  if (bnrClamp > Nn - 1) bnrClamp = Nn - 1;

  const unsigned short* aRow = A + aOff + (long)(rowBase + ar) * lda + ak;

  for (int k0 = 0; k0 < Kd; k0 += 32) {
    { // A tile 128x32 : 2 x uint4 per thread, fully in-bounds by construction
      const unsigned short* src = aRow + k0;
      *(uint4*)&As[ar * 40 + ak]     = *(const uint4*)(src);
      *(uint4*)&As[ar * 40 + ak + 8] = *(const uint4*)(src + 8);
    }
    if (FLAGS & F_TRANSB) {    // Bs[n][k] = Bsrc[n][k]   (contiguous both sides)
      const unsigned short* src = Bsrc + bOff + (long)bnrClamp * ldb + (k0 + bkc);
      *(uint4*)&Bs[bnr * 40 + bkc] = *(const uint4*)src;
    } else {                   // Bs[n][k] = Bsrc[k][n]   (vector load, LDS scatter)
      const unsigned short* src = Bsrc + bOff + (long)(k0 + bkr) * ldb + (colBase + bnc);
      uint4 v = *(const uint4*)src;
      const unsigned short* p = (const unsigned short*)&v;
      #pragma unroll
      for (int j = 0; j < 8; ++j) Bs[(bnc + j) * 40 + bkr] = p[j];
    }
    __syncthreads();

    if (k0 + 32 < Kd) {        // L2 prefetch of next K-slab (global_prefetch_b8)
      __builtin_prefetch(aRow + k0 + 32, 0, 1);
      if (FLAGS & F_TRANSB)
        __builtin_prefetch(Bsrc + bOff + (long)bnrClamp * ldb + (k0 + 32 + bkc), 0, 1);
      else
        __builtin_prefetch(Bsrc + bOff + (long)(k0 + 32 + bkr) * ldb + (colBase + bnc), 0, 1);
    }

    // fragments per CDNA5 ISA VGPR layouts
    union Fr { v16bf v; unsigned int u[8]; } fa0, fa1, fb0, fb1;
    const int am0 = wm * 32 + l15;
    const int am1 = am0 + 16;
    #pragma unroll
    for (int p = 0; p < 8; ++p) {   // A: k = (p/4)*16 + half*8 + 2*(p%4)
      int kk2 = ((p >> 2) << 4) + (half << 3) + ((p & 3) << 1);
      fa0.u[p] = *(const unsigned int*)&As[am0 * 40 + kk2];
      fa1.u[p] = *(const unsigned int*)&As[am1 * 40 + kk2];
    }
    const int bn0 = wn * 32 + l15;
    const int bn1 = bn0 + 16;
    #pragma unroll
    for (int p = 0; p < 8; ++p) {   // B: k = (p/4)*16 + half*8 + 2*(p%4), col = lane%16
      int kk2 = ((p >> 2) << 4) + (half << 3) + ((p & 3) << 1);
      fb0.u[p] = *(const unsigned int*)&Bs[bn0 * 40 + kk2];
      fb1.u[p] = *(const unsigned int*)&Bs[bn1 * 40 + kk2];
    }
    acc00 = __builtin_amdgcn_wmma_f32_16x16x32_bf16(false, fa0.v, false, fb0.v, (short)0, acc00, false, false);
    acc01 = __builtin_amdgcn_wmma_f32_16x16x32_bf16(false, fa0.v, false, fb1.v, (short)0, acc01, false, false);
    acc10 = __builtin_amdgcn_wmma_f32_16x16x32_bf16(false, fa1.v, false, fb0.v, (short)0, acc10, false, false);
    acc11 = __builtin_amdgcn_wmma_f32_16x16x32_bf16(false, fa1.v, false, fb1.v, (short)0, acc11, false, false);
    __syncthreads();
  }

  // epilogue: C/D layout M = r + half*8
  #pragma unroll
  for (int i = 0; i < 2; ++i) {
    #pragma unroll
    for (int r = 0; r < 8; ++r) {
      int gRow = rowBase + wm * 32 + i * 16 + half * 8 + r;
      #pragma unroll
      for (int c = 0; c < 2; ++c) {
        int gCol = colBase + wn * 32 + c * 16 + l15;
        if (gCol >= Nn) continue;
        float v = (i == 0) ? ((c == 0) ? acc00[r] : acc01[r])
                           : ((c == 0) ? acc10[r] : acc11[r]);
        if (FLAGS & F_BIAS) v += bias[gCol];
        if (FLAGS & F_GELU) v = 0.5f * v * (1.0f + erff(v * 0.70710678118654752f));
        long idx;
        if (FLAGS & F_SPLITH) {          // scatter columns into padded per-head layout
          int h2 = gCol / DH;
          int dc = gCol - h2 * DH;
          idx = dOff + (long)h2 * dSH + (long)gRow * ldd + dc;
        } else {
          idx = dOff + (long)gRow * ldd + gCol;
        }
        if (FLAGS & F_RES) v += res[idx];
        if (FLAGS & F_OUTBF16) Dh[idx] = f2bf(v);
        else                   Df[idx] = v;
      }
    }
  }
}

// ---------------------------------------------------------------- unpack
__global__ void unpack_y(const float* __restrict__ yf, float* __restrict__ outp) {
  int tok = blockIdx.x;
  int c   = threadIdx.x;                // 0..95
  int b   = tok / NTOK;
  int n   = tok - b * NTOK;
  int hi  = n / WW, wi = n - hi * WW;
  outp[(((long)b * CCH + c) * HH + hi) * WW + wi] = yf[(long)tok * CCH + c];
}

// ---------------------------------------------------------------- launcher
extern "C" void kernel_launch(void* const* d_in, const int* in_sizes, int n_in,
                              void* d_out, int out_size, void* d_ws, size_t ws_size,
                              hipStream_t stream) {
  (void)in_sizes; (void)n_in; (void)out_size; (void)ws_size;
  const float* detail = (const float*)d_in[0];
  const float* smooth = (const float*)d_in[1];
  const float* ln1_s  = (const float*)d_in[2];
  const float* ln1_b  = (const float*)d_in[3];
  const float* Wq     = (const float*)d_in[4];
  const float* Wk     = (const float*)d_in[5];
  const float* Wv     = (const float*)d_in[6];
  const float* Ek     = (const float*)d_in[7];
  const float* Ev     = (const float*)d_in[8];
  const float* Wo     = (const float*)d_in[9];
  const float* bo     = (const float*)d_in[10];
  const float* ln2_s  = (const float*)d_in[11];
  const float* ln2_b  = (const float*)d_in[12];
  const float* W1     = (const float*)d_in[13];
  const float* b1     = (const float*)d_in[14];
  const float* W2     = (const float*)d_in[15];
  const float* b2     = (const float*)d_in[16];
  const float* Wp     = (const float*)d_in[17];
  const float* bp     = (const float*)d_in[18];
  float* outp = (float*)d_out;

  const long BN = (long)BDIM * NTOK;              // 73728
  char* ws = (char*)d_ws;
  size_t off = 0;
  auto alloc = [&](size_t bytes) -> char* {
    off = (off + 255) & ~(size_t)255;
    char* p = ws + off; off += bytes; return p;
  };
  float*          xf   = (float*)         alloc(BN * DIMF * 4);
  unsigned short* hb   = (unsigned short*)alloc(BN * DIMF * 2);
  unsigned short* qp   = (unsigned short*)alloc((size_t)NHEADS * BN * DHP * 2);   // (H, B*N, 64)
  unsigned short* wqT  = (unsigned short*)alloc((size_t)DIMF * DIMF * 2);
  unsigned short* wkT  = (unsigned short*)alloc((size_t)DIMF * DIMF * 2);
  unsigned short* wvT  = (unsigned short*)alloc((size_t)DIMF * DIMF * 2);
  unsigned short* woT  = (unsigned short*)alloc((size_t)DIMF * DIMF * 2);
  unsigned short* w1T  = (unsigned short*)alloc((size_t)FFD * DIMF * 2);          // (768,192)
  unsigned short* w2T  = (unsigned short*)alloc((size_t)DIMF * FFD * 2);          // (192,768)
  unsigned short* wpT  = (unsigned short*)alloc((size_t)CCH * DIMF * 2);          // (96,192)
  unsigned short* ekT  = (unsigned short*)alloc((size_t)KP * NTOK * 2);           // (256,9216)
  unsigned short* evT  = (unsigned short*)alloc((size_t)KP * NTOK * 2);
  unsigned short* kkp  = (unsigned short*)alloc((size_t)BDIM * NHEADS * KP * DHP * 2); // (B,H,K,64)
  unsigned short* vvp  = (unsigned short*)alloc((size_t)BDIM * NHEADS * KP * DHP * 2);
  float*          dots = (float*)         alloc((size_t)BDIM * NHEADS * NTOK * KP * 4);
  unsigned short* attnb= (unsigned short*)alloc((size_t)BDIM * NHEADS * NTOK * KP * 2);
  // aliases over dead regions
  unsigned short* tkb  = (unsigned short*)dots;                 // dead before dots written
  unsigned short* tvb  = (unsigned short*)dots + BN * DIMF;
  unsigned short* aout = qp;                                    // qp dead after dots
  unsigned short* gbuf = (unsigned short*)dots;                 // dots dead after softmax
  float*          yf   = (float*)attnb;                         // attn dead after attn@vv

  // ---- weight conversion (all weights stored transposed: Bsrc[n][k]) ----
  cvt_transpose_bf16<<<(DIMF*DIMF+255)/256, 256, 0, stream>>>(Wq, wqT, DIMF, DIMF);
  cvt_transpose_bf16<<<(DIMF*DIMF+255)/256, 256, 0, stream>>>(Wk, wkT, DIMF, DIMF);
  cvt_transpose_bf16<<<(DIMF*DIMF+255)/256, 256, 0, stream>>>(Wv, wvT, DIMF, DIMF);
  cvt_transpose_bf16<<<(DIMF*DIMF+255)/256, 256, 0, stream>>>(Wo, woT, DIMF, DIMF);
  cvt_transpose_bf16<<<(DIMF*FFD +255)/256, 256, 0, stream>>>(W1, w1T, DIMF, FFD);
  cvt_transpose_bf16<<<(FFD*DIMF +255)/256, 256, 0, stream>>>(W2, w2T, FFD, DIMF);
  cvt_transpose_bf16<<<(DIMF*CCH +255)/256, 256, 0, stream>>>(Wp, wpT, DIMF, CCH);
  cvt_transpose_bf16<<<(int)(((long)NTOK*KP+255)/256), 256, 0, stream>>>(Ek, ekT, NTOK, KP);
  cvt_transpose_bf16<<<(int)(((long)NTOK*KP+255)/256), 256, 0, stream>>>(Ev, evT, NTOK, KP);

  // ---- zero-fill padded per-head buffers (cols DH..DHP-1 stay zero) ----
  {
    long nq = (long)NHEADS * BN * DHP / 2;                       // u32 count
    fill_zero_u32<<<(int)((nq+255)/256), 256, 0, stream>>>((unsigned int*)qp, nq);
    long nk = (long)BDIM * NHEADS * KP * DHP / 2;
    fill_zero_u32<<<(int)((nk+255)/256), 256, 0, stream>>>((unsigned int*)kkp, nk);
    fill_zero_u32<<<(int)((nk+255)/256), 256, 0, stream>>>((unsigned int*)vvp, nk);
  }

  // ---- x = concat; h = LN1(x) ----
  pack_x<<<(int)BN, DIMF, 0, stream>>>(detail, smooth, xf);
  layernorm_bf16<<<(int)BN, 256, 0, stream>>>(xf, ln1_s, ln1_b, hb);

  dim3 gBN192(3, (unsigned)(BN / 128), 1);
  // q (head-split padded), tk, tv
  gemm_wmma<F_TRANSB|F_OUTBF16|F_SPLITH><<<gBN192, 256, 0, stream>>>(hb, wqT,
      nullptr, nullptr, nullptr, qp,
      (int)BN, DIMF, DIMF, DIMF, DIMF, DHP, 1, 0,0,0,0, 0, BN*DHP);
  gemm_wmma<F_TRANSB|F_OUTBF16><<<gBN192, 256, 0, stream>>>(hb, wkT,
      nullptr, nullptr, nullptr, tkb,
      (int)BN, DIMF, DIMF, DIMF, DIMF, DIMF, 1, 0,0,0,0,0,0);
  gemm_wmma<F_TRANSB|F_OUTBF16><<<gBN192, 256, 0, stream>>>(hb, wvT,
      nullptr, nullptr, nullptr, tvb,
      (int)BN, DIMF, DIMF, DIMF, DIMF, DIMF, 1, 0,0,0,0,0,0);

  // kk = Ek^T @ tk ; vv = Ev^T @ tv  (per batch, head-split padded outputs)
  dim3 gK(3, 2, BDIM);
  gemm_wmma<F_OUTBF16|F_SPLITH><<<gK, 256, 0, stream>>>(ekT, tkb,
      nullptr, nullptr, nullptr, kkp,
      KP, DIMF, NTOK, NTOK, DIMF, DHP, 1,
      0, 0, (long)NTOK*DIMF, 0, (long)NHEADS*KP*DHP, (long)KP*DHP);
  gemm_wmma<F_OUTBF16|F_SPLITH><<<gK, 256, 0, stream>>>(evT, tvb,
      nullptr, nullptr, nullptr, vvp,
      KP, DIMF, NTOK, NTOK, DIMF, DHP, 1,
      0, 0, (long)NTOK*DIMF, 0, (long)NHEADS*KP*DHP, (long)KP*DHP);

  // dots = q @ kk^T   (fully regular: Kd = 64 padded, zeros contribute nothing)
  dim3 gD(4, NTOK/128, BDIM*NHEADS);
  gemm_wmma<F_TRANSB><<<gD, 256, 0, stream>>>(qp, kkp, nullptr, nullptr, dots, nullptr,
      NTOK, KP, DHP, DHP, DHP, KP, NHEADS,
      (long)NTOK*DHP, (long)BN*DHP,
      (long)NHEADS*KP*DHP, (long)KP*DHP,
      (long)NHEADS*NTOK*KP, (long)NTOK*KP);

  softmax_row<<<(unsigned)((long)BDIM*NHEADS*NTOK), 256, 0, stream>>>(dots, attnb);

  // out = attn @ vv   (B reads padded 64 cols; stores guarded to 48)
  dim3 gO(1, NTOK/128, BDIM*NHEADS);
  gemm_wmma<F_OUTBF16><<<gO, 256, 0, stream>>>(attnb, vvp, nullptr, nullptr, nullptr, aout,
      NTOK, DH, KP, KP, DHP, DIMF, NHEADS,
      (long)NHEADS*NTOK*KP, (long)NTOK*KP,
      (long)NHEADS*KP*DHP, (long)KP*DHP,
      (long)NTOK*DIMF, DH);

  // x = x + out @ Wo + bo
  gemm_wmma<F_TRANSB|F_BIAS|F_RES><<<gBN192, 256, 0, stream>>>(aout, woT, bo, xf, xf, nullptr,
      (int)BN, DIMF, DIMF, DIMF, DIMF, DIMF, 1, 0,0,0,0,0,0);

  // FFN
  layernorm_bf16<<<(int)BN, 256, 0, stream>>>(xf, ln2_s, ln2_b, hb);
  dim3 gW1(12, (unsigned)(BN / 128), 1);
  gemm_wmma<F_TRANSB|F_BIAS|F_GELU|F_OUTBF16><<<gW1, 256, 0, stream>>>(hb, w1T, b1,
      nullptr, nullptr, gbuf,
      (int)BN, FFD, DIMF, DIMF, DIMF, FFD, 1, 0,0,0,0,0,0);
  gemm_wmma<F_TRANSB|F_BIAS|F_RES><<<gBN192, 256, 0, stream>>>(gbuf, w2T, b2, xf, xf, nullptr,
      (int)BN, DIMF, FFD, FFD, FFD, DIMF, 1, 0,0,0,0,0,0);

  // y = x @ Wp + bp
  cvt_f32_bf16<<<(int)((BN*DIMF + 255)/256), 256, 0, stream>>>(xf, hb, BN*DIMF);
  dim3 gP(2, (unsigned)(BN / 128), 1);
  gemm_wmma<F_TRANSB|F_BIAS><<<gP, 256, 0, stream>>>(hb, wpT, bp, nullptr, yf, nullptr,
      (int)BN, CCH, DIMF, DIMF, DIMF, CCH, 1, 0,0,0,0,0,0);

  unpack_y<<<(int)BN, CCH, 0, stream>>>(yf, outp);
}
